// SRM_Net_25795573580332
// MI455X (gfx1250) — 8-bit matrix operand order repaired and checked by reading, compiled, not run on hardware
//
#include <hip/hip_runtime.h>
#include <hip/hip_bf16.h>

#define T_NO     200
#define SUB_NO   16
#define E_NO     2000
#define I_NO     500
#define T_DATA   20000
#define NBASIS   16
#define PI_F     3.14159265358979f

typedef __attribute__((ext_vector_type(16))) int   v16i;
typedef __attribute__((ext_vector_type(8)))  float v8f;

typedef int vasync4 __attribute__((vector_size(16)));
#define AS_GLOBAL_V4(p) ((vasync4*)(p))
#define AS_LDS_V4(p)    ((__attribute__((address_space(3))) vasync4*)(p))

__device__ __forceinline__ unsigned pack4_bin_fp8(float4 f) {
  unsigned u0 = __float_as_uint(f.x), u1 = __float_as_uint(f.y);
  unsigned u2 = __float_as_uint(f.z), u3 = __float_as_uint(f.w);
  unsigned lo = __builtin_amdgcn_perm(u1, u0, 0x0C0C0703u);
  unsigned hi = __builtin_amdgcn_perm(u3, u2, 0x07030C0Cu);
  return (lo | hi) & 0x38383838u;
}

template <bool GUARD, int KTOT>
__device__ __forceinline__ v8f slab_wmma(const float* __restrict__ Arow,
                                         const float* __restrict__ Brow,
                                         int k0, int h, v8f acc) {
  v16i a, b;
#pragma unroll
  for (int v = 0; v < 16; ++v) {
    const int w  = v & 7;
    const int ka = k0 + ((v >> 3) << 6) + ((w >> 1) << 4) + ((w & 1) << 2) + (h << 3);
    const int kb = k0 + ((v >> 3) << 6) + ((w >> 1) << 4) + ((w & 1) << 2) + (h << 3);
    unsigned pa = 0u, pb = 0u;
    if (!GUARD || ka < KTOT) pa = pack4_bin_fp8(*(const float4*)(Arow + ka));
    if (!GUARD || kb < KTOT) pb = pack4_bin_fp8(*(const float4*)(Brow + kb));
    a[v] = (int)pa;
    b[v] = (int)pb;
  }
  return __builtin_amdgcn_wmma_f32_16x16x128_fp8_fp8(a, b, (short)0, acc,
                                                     false, false);
}

__global__ __launch_bounds__(32)
void srm_gemm_wmma(const float* __restrict__ Se, const float* __restrict__ Si,
                   const float* __restrict__ Ce, const float* __restrict__ Ci,
                   float* __restrict__ syn_e, float* __restrict__ syn_i) {
  const int lane = threadIdx.x & 31;
  const int h    = lane >> 4;
  const int mr   = lane & 15;
  const int t0   = blockIdx.x * 16;

  const float* __restrict__ Arow_e = Se + (size_t)(t0 + mr) * E_NO;
  const float* __restrict__ Brow_e = Ce + (size_t)mr * E_NO;
  const float* __restrict__ Arow_i = Si + (size_t)(t0 + mr) * I_NO;
  const float* __restrict__ Brow_i = Ci + (size_t)mr * I_NO;

  v8f acc_e = {};
  for (int ks = 0; ks < 15; ++ks) {
    __builtin_prefetch(Arow_e + ks * 128 + 512, 0, 3);
    acc_e = slab_wmma<false, E_NO>(Arow_e, Brow_e, ks * 128, h, acc_e);
  }
  acc_e = slab_wmma<true, E_NO>(Arow_e, Brow_e, 15 * 128, h, acc_e);
#pragma unroll
  for (int r = 0; r < 8; ++r)
    syn_e[(size_t)(t0 + r + (h << 3)) * SUB_NO + mr] = acc_e[r];

  v8f acc_i = {};
  for (int ks = 0; ks < 3; ++ks)
    acc_i = slab_wmma<false, I_NO>(Arow_i, Brow_i, ks * 128, h, acc_i);
  acc_i = slab_wmma<true, I_NO>(Arow_i, Brow_i, 3 * 128, h, acc_i);
#pragma unroll
  for (int r = 0; r < 8; ++r)
    syn_i[(size_t)(t0 + r + (h << 3)) * SUB_NO + mr] = acc_i[r];
}

__global__ __launch_bounds__(256)
void srm_filters(const float* __restrict__ TauE, const float* __restrict__ TauI,
                 const float* __restrict__ We,  const float* __restrict__ Wi,
                 const float* __restrict__ De,  const float* __restrict__ Di,
                 const float* __restrict__ TauS,const float* __restrict__ Wsp,
                 const float* __restrict__ Wh,
                 float* __restrict__ ekern, float* __restrict__ ikern,
                 float* __restrict__ skern, float* __restrict__ hkern,
                 float* __restrict__ outF) {
  for (int idx = threadIdx.x; idx < SUB_NO * T_NO; idx += blockDim.x) {
    const int s = idx / T_NO;
    const int t = idx - s * T_NO;
    const float tf = (float)t;

    float te  = fmaxf(tf - expf(De[s]), 0.f);
    float tte = te / expf(TauE[s]);
    float ek  = tte * expf(-tte) * expf(We[s]);

    float ti  = fmaxf(tf - expf(Di[s]), 0.f);
    float tti = ti / expf(TauI[s]);
    float ik  = -tti * expf(-tti) * expf(Wi[s]);

    float tts = tf / expf(TauS[s]);
    float sk  = tts * expf(-tts) * expf(Wsp[s]);

    float raw = 4.0f * logf(tf + 1.0f);
    float hk  = 0.f;
#pragma unroll
    for (int i = 0; i < NBASIS; ++i) {
      float phi = 0.5f * PI_F * (float)i;
      float b = 0.f;
      if (!(raw < phi - PI_F || raw > phi + PI_F))
        b = 0.5f * __cosf(raw - phi) + 0.5f;
      hk += Wh[s * NBASIS + i] * b;
    }

    ekern[idx] = ek; ikern[idx] = ik; skern[idx] = sk; hkern[idx] = hk;
    outF[(0 * SUB_NO + s) * T_NO + t] = ek;
    outF[(1 * SUB_NO + s) * T_NO + t] = ik;
    outF[(2 * SUB_NO + s) * T_NO + t] = sk;
    outF[(3 * SUB_NO + s) * T_NO + t] = hk;
  }
}

__global__ __launch_bounds__(256)
void srm_conv(const float* __restrict__ syn_e, const float* __restrict__ syn_i,
              const float* __restrict__ ekern, const float* __restrict__ ikern,
              float* __restrict__ syn) {
  __shared__ float se[(T_NO + 16) * SUB_NO];
  __shared__ float si[(T_NO + 16) * SUB_NO];
  __shared__ float ek[SUB_NO * T_NO];
  __shared__ float ik[SUB_NO * T_NO];

  const int tid = threadIdx.x;
  const int n0  = blockIdx.x * 16;

  for (int i = tid; i < SUB_NO * T_NO; i += 256) { ek[i] = ekern[i]; ik[i] = ikern[i]; }

  const int NCHUNK = (T_NO + 16) * SUB_NO / 4;
  for (int i = tid; i < NCHUNK; i += 256) {
    const int row = n0 - T_NO + (i >> 2);
    const int off = i * 4;
    if (row >= 0) {
      __builtin_amdgcn_global_load_async_to_lds_b128(
          AS_GLOBAL_V4(syn_e + (size_t)row * SUB_NO + (i & 3) * 4),
          AS_LDS_V4(se + off), 0, 0);
      __builtin_amdgcn_global_load_async_to_lds_b128(
          AS_GLOBAL_V4(syn_i + (size_t)row * SUB_NO + (i & 3) * 4),
          AS_LDS_V4(si + off), 0, 0);
    } else {
      se[off] = 0.f; se[off + 1] = 0.f; se[off + 2] = 0.f; se[off + 3] = 0.f;
      si[off] = 0.f; si[off + 1] = 0.f; si[off + 2] = 0.f; si[off + 3] = 0.f;
    }
  }
  __builtin_amdgcn_s_wait_asynccnt(0);
  __syncthreads();

  const int s  = tid & 15;
  const int lt = tid >> 4;
  const int rbase = (T_NO - 1 + lt) * SUB_NO + s;
  float acc = 0.f;
#pragma unroll 4
  for (int d = 0; d < T_NO; ++d)
    acc += se[rbase - d * SUB_NO] * ek[s * T_NO + d]
         + si[rbase - d * SUB_NO] * ik[s * T_NO + d];
  syn[(size_t)(n0 + lt) * SUB_NO + s] = acc;
}

__global__ __launch_bounds__(32)
void srm_scan(const float* __restrict__ syn,  const float* __restrict__ hk_g,
              const float* __restrict__ sk_g, const float* __restrict__ Cden,
              const float* __restrict__ Theta, float* __restrict__ spk_out) {
  __shared__ float Ah[T_NO * SUB_NO];
  __shared__ float As[T_NO * SUB_NO];
  __shared__ float hk[SUB_NO * T_NO];
  __shared__ float sk[SUB_NO * T_NO];
  __shared__ float stage[SUB_NO];

  const int lane = threadIdx.x;
  for (int i = lane; i < SUB_NO * T_NO; i += 32) {
    hk[i] = hk_g[i]; sk[i] = sk_g[i]; Ah[i] = 0.f; As[i] = 0.f;
  }
  float cden[SUB_NO];
  float th = 0.f;
  if (lane < SUB_NO) {
    th = Theta[lane];
#pragma unroll
    for (int j = 0; j < SUB_NO; ++j) cden[j] = Cden[lane * SUB_NO + j];
  }
  __syncthreads();

  for (int n = 0; n < T_DATA; ++n) {
    const int ptr = n % T_NO;
    float ah = 0.f, as = 0.f, sv = 0.f;
    if (lane < SUB_NO) {
      ah = Ah[ptr * SUB_NO + lane];
      as = As[ptr * SUB_NO + lane];
      sv = syn[(size_t)n * SUB_NO + lane];
      stage[lane] = as;
    }
    __syncthreads();

    int fire = 0;
    if (lane < SUB_NO) {
      float spk_in = 0.f;
#pragma unroll
      for (int j = 0; j < SUB_NO; ++j) spk_in += cden[j] * stage[j];
      const float sub = ah + spk_in + sv + th;
      fire = (sub > 0.f);
      spk_out[(size_t)n * SUB_NO + lane] = fire ? 1.0f : 0.0f;
      Ah[ptr * SUB_NO + lane] = 0.f;
      As[ptr * SUB_NO + lane] = 0.f;
    }

    unsigned mask = (unsigned)__builtin_amdgcn_ballot_w32(fire != 0) & 0xFFFFu;
    while (mask) {
      const int s2 = __builtin_ctz(mask);
      mask &= mask - 1;
      for (int d = lane; d < T_NO; d += 32) {
        int slot = ptr + 1 + d;
        if (slot >= T_NO) slot -= T_NO;
        Ah[slot * SUB_NO + s2] += hk[s2 * T_NO + d];
        As[slot * SUB_NO + s2] += sk[s2 * T_NO + d];
      }
    }
    __syncthreads();
  }
}

extern "C" void kernel_launch(void* const* d_in, const int* in_sizes, int n_in,
                              void* d_out, int out_size, void* d_ws, size_t ws_size,
                              hipStream_t stream) {
  const float* Se   = (const float*)d_in[0];
  const float* Si   = (const float*)d_in[1];
  const float* Cden = (const float*)d_in[2];
  const float* Ce   = (const float*)d_in[3];
  const float* Ci   = (const float*)d_in[4];
  const float* TauE = (const float*)d_in[5];
  const float* TauI = (const float*)d_in[6];
  const float* We   = (const float*)d_in[7];
  const float* Wi   = (const float*)d_in[8];
  const float* De   = (const float*)d_in[9];
  const float* Di   = (const float*)d_in[10];
  const float* TauS = (const float*)d_in[11];
  const float* Wsp  = (const float*)d_in[12];
  const float* Wh   = (const float*)d_in[13];
  const float* Th   = (const float*)d_in[14];

  float* out = (float*)d_out;
  float* ws  = (float*)d_ws;
  float* syn_e = ws;
  float* syn_i = ws + 320000;
  float* syn   = ws + 640000;
  float* ekern = ws + 960000;
  float* ikern = ws + 963200;
  float* skern = ws + 966400;
  float* hkern = ws + 969600;

  srm_filters<<<1, 256, 0, stream>>>(TauE, TauI, We, Wi, De, Di, TauS, Wsp, Wh,
                                     ekern, ikern, skern, hkern, out + 320000);
  srm_gemm_wmma<<<T_DATA / 16, 32, 0, stream>>>(Se, Si, Ce, Ci, syn_e, syn_i);
  srm_conv<<<T_DATA / 16, 256, 0, stream>>>(syn_e, syn_i, ekern, ikern, syn);
  srm_scan<<<1, 32, 0, stream>>>(syn, hkern, skern, Cden, Th, out);
}
